// GIN_38302518345931
// MI455X (gfx1250) — matrix operand order repaired and checked by reading, not compiled, not run on hardware
//
#include <hip/hip_runtime.h>

// ---------------------------------------------------------------------------
// GINE GNN forward for MI455X (gfx1250, wave32, WMMA bf16 16x16x32, f32 acc)
// GEMMs consume pre-converted bf16 activations + pre-swizzled bf16 weights:
// the inner loop is pure b128/b256 loads + v_wmma (no conversion ALU).
// ---------------------------------------------------------------------------

typedef __attribute__((ext_vector_type(16))) __bf16 v16bf;
typedef __attribute__((ext_vector_type(8)))  __bf16 v8bf;
typedef __attribute__((ext_vector_type(8)))  float  v8f;

constexpr int NN    = 20000;   // nodes
constexpr int NEDGE = 320000;  // edges
constexpr int NGRP  = 512;     // graphs
constexpr int HH    = 256;     // hidden
constexpr int HMLP  = 512;     // MLP hidden
constexpr int D0    = 32;      // conv0 padded input dim (15 -> 32)
constexpr float BN_EPS = 1e-5f;

static __device__ __forceinline__ __bf16 f2bf(float f) {
  // round-to-nearest-even f32 -> bf16
  unsigned u = __builtin_bit_cast(unsigned, f);
  u += 0x7FFFu + ((u >> 16) & 1u);
  unsigned short s = (unsigned short)(u >> 16);
  return __builtin_bit_cast(__bf16, s);
}

// ---------------------------------------------------------------------------
// h0 = concat(x[9], pos_g[3], pos_ex[3]) padded to stride 32 (cols 15..31 = 0)
// ---------------------------------------------------------------------------
__global__ void k_concat(const float* __restrict__ x, const float* __restrict__ pg,
                         const float* __restrict__ pe, float* __restrict__ h0, int n) {
  int i = blockIdx.x * blockDim.x + threadIdx.x;
  if (i >= n) return;
  float* r = h0 + (size_t)i * D0;
#pragma unroll
  for (int j = 0; j < 9; ++j) r[j] = x[(size_t)i * 9 + j];
#pragma unroll
  for (int j = 0; j < 3; ++j) { r[9 + j] = pg[(size_t)i * 3 + j]; r[12 + j] = pe[(size_t)i * 3 + j]; }
#pragma unroll
  for (int j = 15; j < D0; ++j) r[j] = 0.f;
}

__global__ void k_zero(float* __restrict__ p, size_t n) {
  size_t i = (size_t)blockIdx.x * blockDim.x + threadIdx.x;
  if (i < n) p[i] = 0.f;
}

// ---------------------------------------------------------------------------
// Pack W[K<kin, dout] (f32, row-major) into bf16 B-fragment order:
//   idx = (((kb*tilesN + tn)*32 + lane)*16 + e),  K = kb*32 + (e&7) + (lane>>4)*8 + (e>>3)*16,
//   N = tn*16 + (lane&15).  Rows K >= kin are zero (conv0 padding 15 -> 32).
// A lane's 16-element fragment is then one contiguous 32B load in the GEMM.
// ---------------------------------------------------------------------------
__global__ void k_packB(const float* __restrict__ W, __bf16* __restrict__ Wb,
                        int kin, int dout, int total) {
  int i = blockIdx.x * blockDim.x + threadIdx.x;
  if (i >= total) return;
  int e    = i & 15;
  int lane = (i >> 4) & 31;
  int rest = i >> 9;
  int tilesN = dout >> 4;
  int kb = rest / tilesN;
  int tn = rest - kb * tilesN;
  int K = kb * 32 + (e & 7) + (lane >> 4) * 8 + (e >> 3) * 16;
  int N = tn * 16 + (lane & 15);
  Wb[i] = (K < kin) ? f2bf(W[(size_t)K * dout + N]) : f2bf(0.f);
}

// ---------------------------------------------------------------------------
// Edge message + scatter-sum, conv0 (din=15, stride 32)
// ---------------------------------------------------------------------------
__global__ void k_edge15(const int* __restrict__ src, const int* __restrict__ dst,
                         const float* __restrict__ ea, const float* __restrict__ We,
                         const float* __restrict__ be, const float* __restrict__ h,
                         float* __restrict__ agg, int E) {
  int e = blockIdx.x * blockDim.x + threadIdx.x;
  if (e >= E) return;
  float a0 = ea[(size_t)e * 3 + 0], a1 = ea[(size_t)e * 3 + 1], a2 = ea[(size_t)e * 3 + 2];
  int s = src[e], d = dst[e];
#pragma unroll
  for (int j = 0; j < 15; ++j) {
    float v = a0 * We[j] + a1 * We[15 + j] + a2 * We[30 + j] + be[j] + h[(size_t)s * D0 + j];
    v = fmaxf(v, 0.f);
    atomicAdd(&agg[(size_t)d * D0 + j], v);
  }
}

// Edge message + scatter-sum, din=256 (thread = one edge x 8 channels)
__global__ void k_edge256(const int* __restrict__ src, const int* __restrict__ dst,
                          const float* __restrict__ ea, const float* __restrict__ We,
                          const float* __restrict__ be, const float* __restrict__ h,
                          float* __restrict__ agg, int E) {
  int t = blockIdx.x * blockDim.x + threadIdx.x;
  int e = t >> 5;
  int c = (t & 31) * 8;
  if (e >= E) return;
  float a0 = ea[(size_t)e * 3 + 0], a1 = ea[(size_t)e * 3 + 1], a2 = ea[(size_t)e * 3 + 2];
  int s = src[e], d = dst[e];
  const float4* w0 = (const float4*)(We + c);
  const float4* w1 = (const float4*)(We + 256 + c);
  const float4* w2 = (const float4*)(We + 512 + c);
  const float4* bb = (const float4*)(be + c);
  const float4* hh = (const float4*)(h + (size_t)s * 256 + c);
  float m[8];
#pragma unroll
  for (int q = 0; q < 2; ++q) {
    float4 v0 = w0[q], v1 = w1[q], v2 = w2[q], vb = bb[q], vh = hh[q];
    m[q * 4 + 0] = fmaxf(a0 * v0.x + a1 * v1.x + a2 * v2.x + vb.x + vh.x, 0.f);
    m[q * 4 + 1] = fmaxf(a0 * v0.y + a1 * v1.y + a2 * v2.y + vb.y + vh.y, 0.f);
    m[q * 4 + 2] = fmaxf(a0 * v0.z + a1 * v1.z + a2 * v2.z + vb.z + vh.z, 0.f);
    m[q * 4 + 3] = fmaxf(a0 * v0.w + a1 * v1.w + a2 * v2.w + vb.w + vh.w, 0.f);
  }
#pragma unroll
  for (int j = 0; j < 8; ++j) atomicAdd(&agg[(size_t)d * 256 + c + j], m[j]);
}

// z = h + agg, emitted directly in bf16 (GEMM1 is the only consumer of z)
__global__ void k_add_bf16(const float* __restrict__ agg, const float* __restrict__ h,
                           __bf16* __restrict__ z16, size_t n) {
  size_t i = (size_t)blockIdx.x * blockDim.x + threadIdx.x;
  if (i < n) z16[i] = f2bf(agg[i] + h[i]);
}

// ---------------------------------------------------------------------------
// GEMM: C[M,dout] = A16[M,DIN] @ W + bias, optional fused ReLU.
// A16: bf16 row-major. Wb: bf16 pre-swizzled fragment order (see k_packB).
// Wave32 computes an MT x NT grid of 16x16 tiles: per K-step of 32 it issues
// MT A-fragment loads (2 x b128 each), NT B-fragment loads (1 x 32B each) and
// MT*NT v_wmma_f32_16x16x32_bf16 -- no conversion ALU in the loop.
// C/D layout (ISA 7.12.2): lane l -> N = l&15; VGPR r -> M = r + 8*(l>=16).
// ---------------------------------------------------------------------------
template <int DIN, int MT, int NT, bool RELU>
__global__ void k_gemm(const __bf16* __restrict__ A, const __bf16* __restrict__ Wb,
                       const float* __restrict__ bias, float* __restrict__ C,
                       int M, int dout) {
  const int lane = threadIdx.x;                              // 0..31 (wave32)
  const int tilesN = dout >> 4;
  const int tnBlk  = tilesN / NT;
  const int strip = blockIdx.x * blockDim.y + threadIdx.y;   // wave-uniform
  const int sm = strip / tnBlk;
  const int tb = strip - sm * tnBlk;
  if (sm * 16 * MT >= M) return;        // whole-wave exit: EXEC all-ones at WMMA
  const int half = lane >> 4;
  const int l16  = lane & 15;
  const __bf16* __restrict__ Arow[MT];
#pragma unroll
  for (int t = 0; t < MT; ++t)
    Arow[t] = A + (size_t)((sm * MT + t) * 16 + l16) * DIN;
  const __bf16* __restrict__ Bp[NT];
#pragma unroll
  for (int n = 0; n < NT; ++n)
    Bp[n] = Wb + ((size_t)(tb * NT + n) * 32 + lane) * 16;
  v8f acc[MT][NT];
#pragma unroll
  for (int t = 0; t < MT; ++t)
#pragma unroll
    for (int n = 0; n < NT; ++n) acc[t][n] = (v8f){};

  for (int kb = 0; kb < DIN / 32; ++kb) {
    v16bf b[NT];
#pragma unroll
    for (int n = 0; n < NT; ++n)
      b[n] = *(const v16bf*)(Bp[n] + (size_t)kb * tilesN * 512);
    v16bf a[MT];
#pragma unroll
    for (int t = 0; t < MT; ++t) {
      v8bf lo = *(const v8bf*)(Arow[t] + kb * 32 + half * 8);       // K run 0
      v8bf hi = *(const v8bf*)(Arow[t] + kb * 32 + 16 + half * 8);  // K run 1
      a[t] = __builtin_shufflevector(lo, hi, 0, 1, 2, 3, 4, 5, 6, 7,
                                     8, 9, 10, 11, 12, 13, 14, 15);
    }
#pragma unroll
    for (int t = 0; t < MT; ++t)
#pragma unroll
      for (int n = 0; n < NT; ++n)
        acc[t][n] = __builtin_amdgcn_wmma_f32_16x16x32_bf16(
            /*neg_a=*/false, a[t], /*neg_b=*/false, b[n],
            /*c_mod=*/(short)0, acc[t][n], /*reuse_a=*/false, /*reuse_b=*/false);
  }
#pragma unroll
  for (int n = 0; n < NT; ++n) {
    const int colB = (tb * NT + n) * 16 + l16;
    const float bv = bias[colB];
#pragma unroll
    for (int t = 0; t < MT; ++t)
#pragma unroll
      for (int r = 0; r < 8; ++r) {
        float v = acc[t][n][r] + bv;
        if (RELU) v = fmaxf(v, 0.f);
        C[(size_t)((sm * MT + t) * 16 + half * 8 + r) * dout + colB] = v;
      }
  }
}

// ---------------------------------------------------------------------------
// BatchNorm stats: each block scans a contiguous row-chunk row-major
// (coalesced), bins per-column partials in LDS (ds_add_f32), then one global
// atomic per column. cols is a power of two (256 / 512).
// ---------------------------------------------------------------------------
__global__ void k_bnstats(const float* __restrict__ X, int rows, int cols, int rpb,
                          float* __restrict__ sum, float* __restrict__ sumsq) {
  __shared__ float ssum[HMLP], ssq[HMLP];
  for (int c = threadIdx.x; c < cols; c += blockDim.x) { ssum[c] = 0.f; ssq[c] = 0.f; }
  __syncthreads();
  size_t r0 = (size_t)blockIdx.x * rpb;
  if (r0 < (size_t)rows) {
    size_t rend = r0 + rpb; if (rend > (size_t)rows) rend = rows;
    const unsigned mask = (unsigned)cols - 1u;
    for (size_t i = r0 * cols + threadIdx.x, end = rend * cols; i < end; i += blockDim.x) {
      float v = X[i];
      int c = (int)(i & mask);
      atomicAdd(&ssum[c], v);
      atomicAdd(&ssq[c], v * v);
    }
  }
  __syncthreads();
  for (int c = threadIdx.x; c < cols; c += blockDim.x) {
    atomicAdd(&sum[c], ssum[c]);
    atomicAdd(&sumsq[c], ssq[c]);
  }
}

// Fused normalize + affine + ReLU; writes f32 in place and a bf16 copy that
// feeds the next GEMM.
__global__ void k_bnapply_relu(float* __restrict__ X, int rows, int cols,
                               const float* __restrict__ sum, const float* __restrict__ sumsq,
                               const float* __restrict__ g, const float* __restrict__ bt,
                               __bf16* __restrict__ out16) {
  size_t i = (size_t)blockIdx.x * blockDim.x + threadIdx.x;
  if (i >= (size_t)rows * cols) return;
  int c = (int)(i % cols);
  float inv = 1.f / (float)rows;
  float mu  = sum[c] * inv;
  float var = sumsq[c] * inv - mu * mu;     // biased variance (jnp.var)
  float v = (X[i] - mu) * rsqrtf(var + BN_EPS) * g[c] + bt[c];
  v = fmaxf(v, 0.f);
  X[i] = v;
  out16[i] = f2bf(v);
}

// ---------------------------------------------------------------------------
// Pooling: per-group counts + segment-sum of h into pooled[:, off:off+256]
// ---------------------------------------------------------------------------
__global__ void k_count(const int* __restrict__ batch, float* __restrict__ cnt, int n) {
  int i = blockIdx.x * blockDim.x + threadIdx.x;
  if (i < n) atomicAdd(&cnt[batch[i]], 1.f);
}

__global__ void k_pool(const float* __restrict__ h, const int* __restrict__ batch,
                       float* __restrict__ pooled, int n, int off) {
  int t = blockIdx.x * blockDim.x + threadIdx.x;
  int i = t >> 5;
  int c = (t & 31) * 8;
  if (i >= n) return;
  int g = batch[i];
  const float4* hv = (const float4*)(h + (size_t)i * 256 + c);
  float4 v0 = hv[0], v1 = hv[1];
  float m[8] = {v0.x, v0.y, v0.z, v0.w, v1.x, v1.y, v1.z, v1.w};
#pragma unroll
  for (int j = 0; j < 8; ++j)
    atomicAdd(&pooled[(size_t)g * 768 + off + c + j], m[j]);
}

__global__ void k_pooldiv(float* __restrict__ pooled, const float* __restrict__ cnt,
                          __bf16* __restrict__ p16) {
  int i = blockIdx.x * blockDim.x + threadIdx.x;
  if (i >= NGRP * 768) return;
  int g = i / 768;
  float v = pooled[i] / fmaxf(cnt[g], 1.f);
  pooled[i] = v;
  p16[i] = f2bf(v);
}

// Final linear: out[G,2] = Z[G,256] @ W[256,2] + b   (tiny, float4 reads)
__global__ void k_fclast(const float* __restrict__ Z, const float* __restrict__ W,
                         const float* __restrict__ bias, float* __restrict__ out, int G) {
  int i = blockIdx.x * blockDim.x + threadIdx.x;
  if (i >= G * 2) return;
  int g = i >> 1, j = i & 1;
  float s = bias[j];
  const float4* z4 = (const float4*)(Z + (size_t)g * 256);
  for (int k = 0; k < 64; ++k) {
    float4 z = z4[k];
    s += z.x * W[(k * 4 + 0) * 2 + j] + z.y * W[(k * 4 + 1) * 2 + j]
       + z.z * W[(k * 4 + 2) * 2 + j] + z.w * W[(k * 4 + 3) * 2 + j];
  }
  out[i] = s;
}

// ---------------------------------------------------------------------------
// Host orchestration
// ---------------------------------------------------------------------------
static inline int cdiv(long long a, long long b) { return (int)((a + b - 1) / b); }

extern "C" void kernel_launch(void* const* d_in, const int* in_sizes, int n_in,
                              void* d_out, int out_size, void* d_ws, size_t ws_size,
                              hipStream_t stream) {
  (void)in_sizes; (void)n_in; (void)out_size; (void)ws_size;
  // ---- inputs (setup_inputs dict order, params flattened depth-first) ----
  const float* x      = (const float*)d_in[0];
  const float* pos_g  = (const float*)d_in[1];
  const float* pos_ex = (const float*)d_in[2];
  const int*   eidx   = (const int*)d_in[3];
  const int*   src    = eidx;
  const int*   dst    = eidx + NEDGE;
  const float* eattr  = (const float*)d_in[4];
  const int*   batch  = (const int*)d_in[5];
  const float* cv[3][8];  // We, be, W1, b1, g1, bt1, W2, b2
  for (int l = 0; l < 3; ++l)
    for (int p = 0; p < 8; ++p) cv[l][p] = (const float*)d_in[6 + l * 8 + p];
  const float* fc1W  = (const float*)d_in[30];
  const float* fc1b  = (const float*)d_in[31];
  const float* fc1g  = (const float*)d_in[32];
  const float* fc1bt = (const float*)d_in[33];
  const float* fc2W  = (const float*)d_in[34];
  const float* fc2b  = (const float*)d_in[35];
  const float* fc2g  = (const float*)d_in[36];
  const float* fc2bt = (const float*)d_in[37];
  const float* flW   = (const float*)d_in[38];
  const float* flb   = (const float*)d_in[39];
  float* out = (float*)d_out;

  // ---- workspace carve-up ----
  size_t off = 0;
  auto allocB = [&](size_t bytes) -> void* {
    void* p = (char*)d_ws + off;
    off = (off + bytes + 255) & ~(size_t)255;
    return p;
  };
  auto allocF  = [&](size_t n) { return (float*)allocB(n * 4); };
  auto allocH  = [&](size_t n) { return (__bf16*)allocB(n * 2); };

  float* h0     = allocF((size_t)NN * D0);
  float* hA     = allocF((size_t)NN * HH);     // h1 / h3
  float* hB     = allocF((size_t)NN * HH);     // h2
  float* agg    = allocF((size_t)NN * HH);
  float* tbuf   = allocF((size_t)NN * HMLP);   // MLP hidden (f32, pre-BN)
  float* pooled = allocF((size_t)NGRP * 768);
  float* cnt    = allocF(NGRP);
  float* bsum   = allocF(HMLP);
  float* bsq    = allocF(HMLP);
  float* fco1   = allocF((size_t)NGRP * HH);
  float* fco2   = allocF((size_t)NGRP * HH);
  __bf16* z16   = allocH((size_t)NN * HH);     // GEMM1 input (conv0 uses NN*32)
  __bf16* t16   = allocH((size_t)NN * HMLP);   // GEMM2 input (post-BN)
  __bf16* p16   = allocH((size_t)NGRP * 768);  // fc1 input
  __bf16* f16a  = allocH((size_t)NGRP * HH);   // fc2 input (post-BN fc1)
  __bf16* f16b  = allocH((size_t)NGRP * HH);   // post-BN fc2 (bf16 copy unused)
  __bf16* wb0a  = allocH((size_t)D0 * HMLP);   // conv0 W1 (padded 15->32)
  __bf16* wb1a  = allocH((size_t)HH * HMLP);   // conv1 W1
  __bf16* wb2a  = allocH((size_t)HH * HMLP);   // conv2 W1
  __bf16* wb0b  = allocH((size_t)HMLP * HH);   // conv0 W2
  __bf16* wb1b  = allocH((size_t)HMLP * HH);   // conv1 W2
  __bf16* wb2b  = allocH((size_t)HMLP * HH);   // conv2 W2
  __bf16* wbf1  = allocH((size_t)768 * HH);    // fc1 W
  __bf16* wbf2  = allocH((size_t)HH * HH);     // fc2 W

  const dim3 gb(32, 4);  // 4 wave32s per block for GEMM
  constexpr int MT = 2, NT = 2;

  auto packW = [&](const float* W, __bf16* Wb, int kin, int kpad, int dout) {
    int total = kpad * dout;
    k_packB<<<cdiv(total, 256), 256, 0, stream>>>(W, Wb, kin, dout, total);
  };
  auto gemm_grid = [&](int M, int dout) {
    return cdiv((long long)(M / (16 * MT)) * ((dout / 16) / NT), 4);
  };
  auto run_bn = [&](float* X, int rows, int cols, const float* g, const float* bt, __bf16* o16) {
    int nblk = (rows > 4096) ? 128 : 8;
    k_zero<<<cdiv(2 * HMLP, 256), 256, 0, stream>>>(bsum, (size_t)2 * HMLP);  // bsum+bsq adjacent
    k_bnstats<<<nblk, 256, 0, stream>>>(X, rows, cols, cdiv(rows, nblk), bsum, bsq);
    k_bnapply_relu<<<cdiv((long long)rows * cols, 256), 256, 0, stream>>>(X, rows, cols, bsum, bsq, g, bt, o16);
  };

  // ---- weight pre-pack (bf16, B-fragment order) ----
  packW(cv[0][2], wb0a, 15,  D0,   HMLP);
  packW(cv[1][2], wb1a, HH,  HH,   HMLP);
  packW(cv[2][2], wb2a, HH,  HH,   HMLP);
  packW(cv[0][6], wb0b, HMLP, HMLP, HH);
  packW(cv[1][6], wb1b, HMLP, HMLP, HH);
  packW(cv[2][6], wb2b, HMLP, HMLP, HH);
  packW(fc1W,     wbf1, 768, 768,  HH);
  packW(fc2W,     wbf2, HH,  HH,   HH);

  // ---- h0 + pooling accumulators ----
  k_concat<<<cdiv(NN, 256), 256, 0, stream>>>(x, pos_g, pos_ex, h0, NN);
  k_zero<<<cdiv(NGRP * 768, 256), 256, 0, stream>>>(pooled, (size_t)NGRP * 768);
  k_zero<<<cdiv(NGRP, 256), 256, 0, stream>>>(cnt, NGRP);
  k_count<<<cdiv(NN, 256), 256, 0, stream>>>(batch, cnt, NN);

  // ---- conv0 (din padded to 32) ----
  k_zero<<<cdiv((long long)NN * D0, 256), 256, 0, stream>>>(agg, (size_t)NN * D0);
  k_edge15<<<cdiv(NEDGE, 256), 256, 0, stream>>>(src, dst, eattr, cv[0][0], cv[0][1], h0, agg, NEDGE);
  k_add_bf16<<<cdiv((long long)NN * D0, 256), 256, 0, stream>>>(agg, h0, z16, (size_t)NN * D0);
  k_gemm<D0, MT, NT, false><<<gemm_grid(NN, HMLP), gb, 0, stream>>>(z16, wb0a, cv[0][3], tbuf, NN, HMLP);
  run_bn(tbuf, NN, HMLP, cv[0][4], cv[0][5], t16);
  k_gemm<HMLP, MT, NT, true><<<gemm_grid(NN, HH), gb, 0, stream>>>(t16, wb0b, cv[0][7], hA, NN, HH);
  k_pool<<<cdiv((long long)NN * 32, 256), 256, 0, stream>>>(hA, batch, pooled, NN, 0);

  // ---- conv1 ----
  k_zero<<<cdiv((long long)NN * HH, 256), 256, 0, stream>>>(agg, (size_t)NN * HH);
  k_edge256<<<cdiv((long long)NEDGE * 32, 256), 256, 0, stream>>>(src, dst, eattr, cv[1][0], cv[1][1], hA, agg, NEDGE);
  k_add_bf16<<<cdiv((long long)NN * HH, 256), 256, 0, stream>>>(agg, hA, z16, (size_t)NN * HH);
  k_gemm<HH, MT, NT, false><<<gemm_grid(NN, HMLP), gb, 0, stream>>>(z16, wb1a, cv[1][3], tbuf, NN, HMLP);
  run_bn(tbuf, NN, HMLP, cv[1][4], cv[1][5], t16);
  k_gemm<HMLP, MT, NT, true><<<gemm_grid(NN, HH), gb, 0, stream>>>(t16, wb1b, cv[1][7], hB, NN, HH);
  k_pool<<<cdiv((long long)NN * 32, 256), 256, 0, stream>>>(hB, batch, pooled, NN, 256);

  // ---- conv2 ----
  k_zero<<<cdiv((long long)NN * HH, 256), 256, 0, stream>>>(agg, (size_t)NN * HH);
  k_edge256<<<cdiv((long long)NEDGE * 32, 256), 256, 0, stream>>>(src, dst, eattr, cv[2][0], cv[2][1], hB, agg, NEDGE);
  k_add_bf16<<<cdiv((long long)NN * HH, 256), 256, 0, stream>>>(agg, hB, z16, (size_t)NN * HH);
  k_gemm<HH, MT, NT, false><<<gemm_grid(NN, HMLP), gb, 0, stream>>>(z16, wb2a, cv[2][3], tbuf, NN, HMLP);
  run_bn(tbuf, NN, HMLP, cv[2][4], cv[2][5], t16);
  k_gemm<HMLP, MT, NT, true><<<gemm_grid(NN, HH), gb, 0, stream>>>(t16, wb2b, cv[2][7], hA, NN, HH);
  k_pool<<<cdiv((long long)NN * 32, 256), 256, 0, stream>>>(hA, batch, pooled, NN, 512);

  // ---- mean pool + FC head ----
  k_pooldiv<<<cdiv(NGRP * 768, 256), 256, 0, stream>>>(pooled, cnt, p16);
  k_gemm<768, MT, NT, false><<<gemm_grid(NGRP, HH), gb, 0, stream>>>(p16, wbf1, fc1b, fco1, NGRP, HH);
  run_bn(fco1, NGRP, HH, fc1g, fc1bt, f16a);
  k_gemm<HH, MT, NT, false><<<gemm_grid(NGRP, HH), gb, 0, stream>>>(f16a, wbf2, fc2b, fco2, NGRP, HH);
  run_bn(fco2, NGRP, HH, fc2g, fc2bt, f16b);
  k_fclast<<<cdiv(NGRP * 2, 256), 256, 0, stream>>>(fco2, flW, flb, out, NGRP);
}
